// RowColumnAttention_39238821216522
// MI455X (gfx1250) — matrix operand order repaired and checked by reading, not compiled, not run on hardware
//
#include <hip/hip_runtime.h>
#include <math.h>

// ---------------- problem constants ----------------
#define C_DIM 1024
#define HEADS 16
#define KVH   8
#define HD    64
#define INTER 4096
#define BB    4
#define RR    32
#define TT    16
#define LL    112
#define SROW  128            // T + L
#define NEGF  (-3.40282346638528859812e+38f)
#define LOG_THETA 13.815510557964274f   // ln(1e6)

typedef __attribute__((ext_vector_type(16))) __bf16 v16bf;
typedef __attribute__((ext_vector_type(8)))  float  v8f;

static __device__ __forceinline__ unsigned short f2bf(float f) {
  unsigned int u = __float_as_uint(f);
  u += 0x7fffu + ((u >> 16) & 1u);          // round-to-nearest-even
  return (unsigned short)(u >> 16);
}

// ---------------- f32 -> bf16 convert ----------------
__global__ __launch_bounds__(256) void k_cvt_bf16(const float* __restrict__ x,
    unsigned short* __restrict__ y, int n) {
  for (int i = blockIdx.x * blockDim.x + threadIdx.x; i < n; i += gridDim.x * blockDim.x)
    y[i] = f2bf(x[i]);
}

// ---------------- NT GEMM via WMMA bf16: Y[m,n] (+)= sum_k X[m,k]*W[n,k] ----------------
// X: (M,K) bf16, W: (N,K) bf16, Y: (M,N) f32.
// Wave tile = 32x64 (2 A-tiles x 4 B-tiles, 8 accumulators) -> 24 B loaded per WMMA.
// Block = 8 waves arranged 2(M) x 4(N) => 64x256 block tile. M%64==0, N%256==0 here.
__global__ __launch_bounds__(256) void k_gemm_nt(const unsigned short* __restrict__ X,
    const unsigned short* __restrict__ W, float* __restrict__ Y,
    int M, int N, int K, int acc) {
  const int lane = threadIdx.x & 31;
  const int wave = threadIdx.x >> 5;
  const int half = lane >> 4;          // lane group 0/1
  const int row  = lane & 15;
  const int m0 = blockIdx.y * 64 + (wave >> 2) * 32;   // 2 waves in M
  const int n0 = blockIdx.x * 256 + (wave & 3) * 64;   // 4 waves in N
  if (m0 >= M || n0 >= N) return;      // wave-uniform

  v8f c[2][4];
  if (acc) {
#pragma unroll
    for (int i = 0; i < 2; ++i)
#pragma unroll
      for (int t = 0; t < 4; ++t)
#pragma unroll
        for (int r = 0; r < 8; ++r)
          c[i][t][r] = Y[(size_t)(m0 + i * 16 + half * 8 + r) * N + n0 + t * 16 + row];
  } else {
#pragma unroll
    for (int i = 0; i < 2; ++i)
#pragma unroll
      for (int t = 0; t < 4; ++t)
#pragma unroll
        for (int r = 0; r < 8; ++r) c[i][t][r] = 0.0f;
  }

  const unsigned short* xp0 = X + (size_t)(m0 + row) * K;        // A tile 0 (rows m0..)
  const unsigned short* xp1 = X + (size_t)(m0 + 16 + row) * K;   // A tile 1
  const unsigned short* wp[4];
#pragma unroll
  for (int t = 0; t < 4; ++t) wp[t] = W + (size_t)(n0 + t * 16 + row) * K;

  union U { v16bf v; unsigned int u[8]; };
  for (int k0 = 0; k0 < K; k0 += 32) {
    U a[2], b[4];
#pragma unroll
    for (int p = 0; p < 8; ++p) {
      // A 16x32 bf16 layout: lanes0-15 K=0..7,16..23 ; lanes16-31 K=8..15,24..31
      const int ka = half * 8 + 2 * p + ((p >= 4) ? 8 : 0);
      a[0].u[p] = *(const unsigned int*)(xp0 + k0 + ka);
      a[1].u[p] = *(const unsigned int*)(xp1 + k0 + ka);
      // B 32x16 bf16 layout: lanes0-15 K=0..7,16..23 ; lanes16-31 K=8..15,24..31
      const int kb = half * 8 + 2 * p + ((p >= 4) ? 8 : 0);
      b[0].u[p] = *(const unsigned int*)(wp[0] + k0 + kb);
      b[1].u[p] = *(const unsigned int*)(wp[1] + k0 + kb);
      b[2].u[p] = *(const unsigned int*)(wp[2] + k0 + kb);
      b[3].u[p] = *(const unsigned int*)(wp[3] + k0 + kb);
    }
    // prefetch next K block (speculative: OOB prefetch is silently dropped)
    __builtin_prefetch(xp0 + k0 + 32, 0, 1);
    __builtin_prefetch(xp1 + k0 + 32, 0, 1);
    __builtin_prefetch(wp[0] + k0 + 32, 0, 1);
    __builtin_prefetch(wp[2] + k0 + 32, 0, 1);
#pragma unroll
    for (int i = 0; i < 2; ++i)
#pragma unroll
      for (int t = 0; t < 4; ++t)
        c[i][t] = __builtin_amdgcn_wmma_f32_16x16x32_bf16(false, a[i].v, false, b[t].v,
                                                          (short)0, c[i][t], false, false);
  }
#pragma unroll
  for (int i = 0; i < 2; ++i)
#pragma unroll
    for (int t = 0; t < 4; ++t)
#pragma unroll
      for (int r = 0; r < 8; ++r)
        Y[(size_t)(m0 + i * 16 + half * 8 + r) * N + n0 + t * 16 + row] = c[i][t][r];
}

// ---------------- RMSNorm (C=1024) -> bf16 ----------------
__global__ __launch_bounds__(256) void k_rmsnorm(const float* __restrict__ x,
    const float* __restrict__ w, unsigned short* __restrict__ y) {
  __shared__ float red[256];
  const size_t base = (size_t)blockIdx.x * C_DIM;
  float v[4]; float s = 0.0f;
#pragma unroll
  for (int i = 0; i < 4; ++i) { v[i] = x[base + threadIdx.x + 256 * i]; s += v[i] * v[i]; }
  red[threadIdx.x] = s; __syncthreads();
  for (int off = 128; off > 0; off >>= 1) {
    if ((int)threadIdx.x < off) red[threadIdx.x] += red[threadIdx.x + off];
    __syncthreads();
  }
  const float scale = rsqrtf(red[0] * (1.0f / 1024.0f) + 1e-6f);
#pragma unroll
  for (int i = 0; i < 4; ++i) {
    const int cc = threadIdx.x + 256 * i;
    y[base + cc] = f2bf(v[i] * scale * w[cc]);
  }
}

// ---------------- per-head RMSNorm + RoPE, in place on f32 (tokens, nh*64) ----------------
__global__ __launch_bounds__(64) void k_qknorm_rope(float* __restrict__ x,
    const float* __restrict__ w, int S, int nh) {
  __shared__ float red[64];
  __shared__ float sx[64];
  const int tok = blockIdx.x, h = blockIdx.y, d = threadIdx.x;
  const size_t base = (size_t)tok * (nh * HD) + h * HD;
  const float v = x[base + d];
  red[d] = v * v; __syncthreads();
  for (int off = 32; off > 0; off >>= 1) { if (d < off) red[d] += red[d + off]; __syncthreads(); }
  const float scale = rsqrtf(red[0] * (1.0f / 64.0f) + 1e-6f);
  const float xv = v * scale * w[d];
  sx[d] = xv; __syncthreads();
  const float rot = (d < 32) ? -sx[d + 32] : sx[d - 32];
  const int pos = tok % S;
  const float ang = (float)pos * __expf(-LOG_THETA * ((float)(d & 31) * (1.0f / 32.0f)));
  x[base + d] = xv * cosf(ang) + rot * sinf(ang);
}

// ---------------- attention: one block per (seq, head), online softmax ----------------
__global__ __launch_bounds__(128) void k_attn(const float* __restrict__ Q,
    const float* __restrict__ K, const float* __restrict__ V,
    const float* __restrict__ mask, unsigned short* __restrict__ O, int S) {
  __shared__ float Ks[SROW][HD];
  __shared__ float Vs[SROW][HD];
  const int seq = blockIdx.x, h = blockIdx.y, t = threadIdx.x;
  const int kvh = h >> 1;                       // HEADS/KV_HEADS = 2
  const int total = S * HD;
  for (int i = t; i < total; i += blockDim.x) {
    const int kt = i >> 6, d = i & 63;
    const size_t src = (size_t)(seq * S + kt) * (KVH * HD) + kvh * HD + d;
    Ks[kt][d] = K[src];
    Vs[kt][d] = V[src];
  }
  __syncthreads();
  float qr[HD];
  const size_t qbase = (size_t)(seq * S + t) * (HEADS * HD) + h * HD;
#pragma unroll
  for (int d = 0; d < HD; ++d) qr[d] = Q[qbase + d];
  const float* mrow = mask + ((size_t)seq * S + t) * S;
  float m = -INFINITY, l = 0.0f;
  float acc[HD];
#pragma unroll
  for (int d = 0; d < HD; ++d) acc[d] = 0.0f;
  for (int kt = 0; kt < S; ++kt) {
    float s = 0.0f;
#pragma unroll
    for (int d = 0; d < HD; ++d) s += qr[d] * Ks[kt][d];
    s = s * 0.125f + mrow[kt];                  // HEAD_DIM^-0.5
    const float mn = fmaxf(m, s);
    const float corr = __expf(m - mn);
    const float p = __expf(s - mn);
    l = l * corr + p;
#pragma unroll
    for (int d = 0; d < HD; ++d) acc[d] = acc[d] * corr + p * Vs[kt][d];
    m = mn;
  }
  const float inv = 1.0f / l;
#pragma unroll
  for (int d = 0; d < HD; ++d) O[qbase + d] = f2bf(acc[d] * inv);
}

// ---------------- SwiGLU combine ----------------
__global__ __launch_bounds__(256) void k_silu_mul(const float* __restrict__ g,
    const float* __restrict__ u, unsigned short* __restrict__ a, int n) {
  const int i = blockIdx.x * 256 + threadIdx.x;
  if (i < n) { const float x = g[i]; a[i] = f2bf((x / (1.0f + __expf(-x))) * u[i]); }
}

// ---------------- layout shuffles ----------------
__global__ void k_concat(const float* __restrict__ learned, const float* __restrict__ dict,
                         float* __restrict__ h) {
  const int rowid = blockIdx.x;                 // br*128 + s
  const int s = rowid & (SROW - 1), br = rowid >> 7;
  const float* src = (s < TT) ? (learned + ((size_t)br * TT + s) * C_DIM)
                              : (dict + ((size_t)br * LL + (s - TT)) * C_DIM);
  float* dst = h + (size_t)rowid * C_DIM;
  for (int i = threadIdx.x; i < C_DIM; i += 256) dst[i] = src[i];
}

__global__ void k_build_hcol(const float* __restrict__ hrow, float* __restrict__ hcol) {
  const int rowid = blockIdx.x;                 // (b*L + l)*R + r
  const int r = rowid & 31;
  const int bl = rowid >> 5;
  const int l = bl % LL, b = bl / LL;
  const float* src = hrow + ((size_t)(b * RR + r) * SROW + TT + l) * C_DIM;
  float* dst = hcol + (size_t)rowid * C_DIM;
  for (int i = threadIdx.x; i < C_DIM; i += 256) dst[i] = src[i];
}

__global__ void k_out_learned(const float* __restrict__ hrow, float* __restrict__ out) {
  const int rowid = blockIdx.x;                 // br*16 + t
  const int t = rowid & 15, br = rowid >> 4;
  const float* src = hrow + ((size_t)br * SROW + t) * C_DIM;
  float* dst = out + (size_t)rowid * C_DIM;
  for (int i = threadIdx.x; i < C_DIM; i += 256) dst[i] = src[i];
}

__global__ void k_out_dict(const float* __restrict__ hcol, float* __restrict__ out) {
  const int rowid = blockIdx.x;                 // (b*R + r)*L + l
  const int l = rowid % LL;
  const int br = rowid / LL;
  const int b = br >> 5, r = br & 31;
  const float* src = hcol + ((size_t)(b * LL + l) * RR + r) * C_DIM;
  float* dst = out + (size_t)rowid * C_DIM;
  for (int i = threadIdx.x; i < C_DIM; i += 256) dst[i] = src[i];
}

// ---------------- mask builders (bool inputs assumed 1 byte / element) ----------------
__global__ void k_row_mask(const unsigned char* __restrict__ dpad,
                           const unsigned char* __restrict__ rpad, float* __restrict__ mask) {
  const int rowflat = blockIdx.x;               // n*128 + i
  const int i = rowflat & (SROW - 1), n = rowflat >> 7;
  const int j = threadIdx.x;
  float v = 0.0f;
  if (j >= TT) {
    if (i >= TT) v = (j <= i) ? 0.0f : NEGF;    // causal block overwrites pad
    else        v = (dpad[n * LL + (j - TT)] | rpad[n]) ? NEGF : 0.0f;
  }
  mask[(size_t)rowflat * SROW + j] = v;
}

__global__ void k_col_mask(const unsigned char* __restrict__ rpad, float* __restrict__ mask) {
  const int rowflat = blockIdx.x;               // (b*L + l)*R + i
  const int n = rowflat >> 5;
  const int b = n / LL;
  const int j = threadIdx.x;
  mask[(size_t)rowflat * RR + j] = rpad[b * RR + j] ? NEGF : 0.0f;
}

// ---------------- host side ----------------
struct FP  { const float *qw,*kw,*vw,*ow,*gw,*uw,*dw,*iln,*pln,*qn,*kn; };
struct BFW { unsigned short *qw,*kw,*vw,*ow,*gw,*uw,*dw; const float *iln,*pln,*qn,*kn; };

static FP map_params(void* const* din, const int* sz, int base) {
  const float* a[11];
  for (int i = 0; i < 11; ++i) a[i] = (const float*)din[base + i];
  FP p;
  if (sz[base] == 1048576 && sz[base + 4] == 4194304) {
    // insertion order: q_w,k_w,v_w,o_w,gate_w,up_w,down_w,input_ln,post_ln,q_norm,k_norm
    p.qw=a[0]; p.kw=a[1]; p.vw=a[2]; p.ow=a[3]; p.gw=a[4]; p.uw=a[5]; p.dw=a[6];
    p.iln=a[7]; p.pln=a[8]; p.qn=a[9]; p.kn=a[10];
  } else {
    // alphabetical (jax tree order): down,gate,input_ln,k_norm,k_w,o_w,post_ln,q_norm,q_w,up_w,v_w
    p.dw=a[0]; p.gw=a[1]; p.iln=a[2]; p.kn=a[3]; p.kw=a[4]; p.ow=a[5];
    p.pln=a[6]; p.qn=a[7]; p.qw=a[8]; p.uw=a[9]; p.vw=a[10];
  }
  return p;
}

static void cvt(const float* s, unsigned short* d, int n, hipStream_t st) {
  int blocks = (n + 255) / 256; if (blocks > 8192) blocks = 8192;
  k_cvt_bf16<<<blocks, 256, 0, st>>>(s, d, n);
}

static void run_layer(float* h, int nseq, int S, const BFW& L, const float* mask,
                      unsigned short* xn, float* qb, float* kb, float* vb,
                      unsigned short* ob, float* gb, float* ub, unsigned short* actb,
                      hipStream_t st) {
  const int tokens = nseq * S;                        // multiple of 2048
  k_rmsnorm<<<tokens, 256, 0, st>>>(h, L.iln, xn);
  k_gemm_nt<<<dim3(C_DIM / 256, tokens / 64), 256, 0, st>>>(xn, L.qw, qb, tokens, C_DIM, C_DIM, 0);
  k_gemm_nt<<<dim3(512 / 256,  tokens / 64), 256, 0, st>>>(xn, L.kw, kb, tokens, 512,  C_DIM, 0);
  k_gemm_nt<<<dim3(512 / 256,  tokens / 64), 256, 0, st>>>(xn, L.vw, vb, tokens, 512,  C_DIM, 0);
  k_qknorm_rope<<<dim3(tokens, HEADS), 64, 0, st>>>(qb, L.qn, S, HEADS);
  k_qknorm_rope<<<dim3(tokens, KVH),   64, 0, st>>>(kb, L.kn, S, KVH);
  k_attn<<<dim3(nseq, HEADS), S, 0, st>>>(qb, kb, vb, mask, ob, S);
  k_gemm_nt<<<dim3(C_DIM / 256, tokens / 64), 256, 0, st>>>(ob, L.ow, h, tokens, C_DIM, C_DIM, 1);
  k_rmsnorm<<<tokens, 256, 0, st>>>(h, L.pln, xn);
  const int CH = 2048;                                // MLP token chunk
  for (int m0 = 0; m0 < tokens; m0 += CH) {
    const unsigned short* xc = xn + (size_t)m0 * C_DIM;
    k_gemm_nt<<<dim3(INTER / 256, CH / 64), 256, 0, st>>>(xc, L.gw, gb, CH, INTER, C_DIM, 0);
    k_gemm_nt<<<dim3(INTER / 256, CH / 64), 256, 0, st>>>(xc, L.uw, ub, CH, INTER, C_DIM, 0);
    k_silu_mul<<<(CH * INTER) / 256, 256, 0, st>>>(gb, ub, actb, CH * INTER);
    k_gemm_nt<<<dim3(C_DIM / 256, CH / 64), 256, 0, st>>>(actb, L.dw,
        h + (size_t)m0 * C_DIM, CH, C_DIM, INTER, 1);
  }
}

extern "C" void kernel_launch(void* const* d_in, const int* in_sizes, int n_in,
                              void* d_out, int out_size, void* d_ws, size_t ws_size,
                              hipStream_t stream) {
  (void)n_in; (void)out_size; (void)ws_size;
  const float* learned  = (const float*)d_in[0];
  const float* dict_emb = (const float*)d_in[1];
  const unsigned char* dpad = (const unsigned char*)d_in[2];
  const unsigned char* rpad = (const unsigned char*)d_in[3];
  FP rp = map_params(d_in, in_sizes, 4);
  FP cp = map_params(d_in, in_sizes, 15);

  // ---- workspace carve ----
  char* base = (char*)d_ws; size_t off = 0;
  auto take = [&](size_t nbytes) -> void* {
    void* p = base + off; off += (nbytes + 255) & ~(size_t)255; return p;
  };
  const size_t TOK_R = (size_t)BB * RR * SROW;        // 16384
  const size_t TOK_C = (size_t)BB * LL * RR;          // 14336
  BFW Lr, Lc;
  Lr.qw = (unsigned short*)take(1048576 * 2); Lr.kw = (unsigned short*)take(524288 * 2);
  Lr.vw = (unsigned short*)take(524288 * 2);  Lr.ow = (unsigned short*)take(1048576 * 2);
  Lr.gw = (unsigned short*)take(4194304 * 2); Lr.uw = (unsigned short*)take(4194304 * 2);
  Lr.dw = (unsigned short*)take(4194304 * 2);
  Lc.qw = (unsigned short*)take(1048576 * 2); Lc.kw = (unsigned short*)take(524288 * 2);
  Lc.vw = (unsigned short*)take(524288 * 2);  Lc.ow = (unsigned short*)take(1048576 * 2);
  Lc.gw = (unsigned short*)take(4194304 * 2); Lc.uw = (unsigned short*)take(4194304 * 2);
  Lc.dw = (unsigned short*)take(4194304 * 2);
  Lr.iln = rp.iln; Lr.pln = rp.pln; Lr.qn = rp.qn; Lr.kn = rp.kn;
  Lc.iln = cp.iln; Lc.pln = cp.pln; Lc.qn = cp.qn; Lc.kn = cp.kn;

  float*          hrow = (float*)take(TOK_R * C_DIM * 4);
  float*          hcol = (float*)take(TOK_C * C_DIM * 4);
  unsigned short* xn   = (unsigned short*)take(TOK_R * C_DIM * 2);
  float*          qb   = (float*)take(TOK_R * C_DIM * 4);
  float*          kb   = (float*)take(TOK_R * 512 * 4);
  float*          vb   = (float*)take(TOK_R * 512 * 4);
  unsigned short* ob   = (unsigned short*)take(TOK_R * C_DIM * 2);
  float*          maskb= (float*)take((size_t)BB * RR * SROW * SROW * 4); // covers col mask too
  float*          gb   = (float*)take((size_t)2048 * INTER * 4);
  float*          ub   = (float*)take((size_t)2048 * INTER * 4);
  unsigned short* actb = (unsigned short*)take((size_t)2048 * INTER * 2);

  // ---- convert weights to bf16 ----
  cvt(rp.qw, Lr.qw, 1048576, stream); cvt(rp.kw, Lr.kw, 524288, stream);
  cvt(rp.vw, Lr.vw, 524288, stream);  cvt(rp.ow, Lr.ow, 1048576, stream);
  cvt(rp.gw, Lr.gw, 4194304, stream); cvt(rp.uw, Lr.uw, 4194304, stream);
  cvt(rp.dw, Lr.dw, 4194304, stream);
  cvt(cp.qw, Lc.qw, 1048576, stream); cvt(cp.kw, Lc.kw, 524288, stream);
  cvt(cp.vw, Lc.vw, 524288, stream);  cvt(cp.ow, Lc.ow, 1048576, stream);
  cvt(cp.gw, Lc.gw, 4194304, stream); cvt(cp.uw, Lc.uw, 4194304, stream);
  cvt(cp.dw, Lc.dw, 4194304, stream);

  // ---- row pass: 128 seqs x 128 tokens ----
  k_concat<<<(int)TOK_R, 256, 0, stream>>>(learned, dict_emb, hrow);
  k_row_mask<<<(int)TOK_R, SROW, 0, stream>>>(dpad, rpad, maskb);
  run_layer(hrow, BB * RR, SROW, Lr, maskb, xn, qb, kb, vb, ob, gb, ub, actb, stream);

  // ---- column pass: 448 seqs x 32 tokens ----
  k_build_hcol<<<(int)TOK_C, 256, 0, stream>>>(hrow, hcol);
  k_col_mask<<<(int)TOK_C, RR, 0, stream>>>(rpad, maskb);
  run_layer(hcol, BB * LL, RR, Lc, maskb, xn, qb, kb, vb, ob, gb, ub, actb, stream);

  // ---- outputs: learned_upd (B,R,T,C) then dict_out (B,R,L,C) ----
  float* out = (float*)d_out;
  k_out_learned<<<BB * RR * TT, 256, 0, stream>>>(hrow, out);
  k_out_dict<<<BB * RR * LL, 256, 0, stream>>>(hcol, out + (size_t)BB * RR * TT * C_DIM);
}
